// GlobalMixture_68307159875840
// MI455X (gfx1250) — matrix operand order repaired and checked by reading, not compiled, not run on hardware
//
#include <hip/hip_runtime.h>
#include <hip/hip_bf16.h>

#define B_  16
#define L_  8192
#define D_  256
#define BK  32
#define LDK 40   // padded LDS row stride (ushorts): 80B, multiple of 16B

typedef __attribute__((ext_vector_type(16))) _Float16 v16h;
typedef __attribute__((ext_vector_type(8)))  _Float16 v8h;
typedef __attribute__((ext_vector_type(2)))  _Float16 v2h;
typedef __attribute__((ext_vector_type(8)))  float    v8f;
typedef unsigned short ush;

// Pack two f32 -> two f16 in one dword (elem0 low, elem1 high).
__device__ __forceinline__ unsigned pk_f16(float lo, float hi) {
#if __has_builtin(__builtin_amdgcn_cvt_pkrtz)
  auto r = __builtin_amdgcn_cvt_pkrtz(lo, hi);   // v_cvt_pk_rtz_f16_f32 (__fp16 x2)
  return __builtin_bit_cast(unsigned, r);
#else
  v2h r; r.x = (_Float16)lo; r.y = (_Float16)hi;
  return __builtin_bit_cast(unsigned, r);
#endif
}

union V16 { v16h v; v8h h[2]; };

__device__ __forceinline__ v8f wmma_f16(v16h a, v16h b, v8f c) {
  return __builtin_amdgcn_wmma_f32_16x16x32_f16(
      /*neg_a=*/false, a, /*neg_b=*/false, b,
      /*c_mod=*/(short)0, c, /*reuse_a=*/false, /*reuse_b=*/false);
}

// ---------------- Kernel 1: per-token L2 norms (one wave32 per token) -------
__global__ void norms_kernel(const float* __restrict__ x, float* __restrict__ norms) {
  int wave = (blockIdx.x * blockDim.x + threadIdx.x) >> 5;
  int lane = threadIdx.x & 31;
  const float* p = x + (size_t)wave * D_;
  float s = 0.f;
  #pragma unroll
  for (int j = 0; j < 8; ++j) {
    float v = p[lane + 32 * j];
    s = fmaf(v, v, s);
  }
  #pragma unroll
  for (int o = 16; o > 0; o >>= 1) s += __shfl_xor(s, o, 32);
  if (lane == 0) norms[wave] = sqrtf(s);
}

// ------- Kernel 2: sqrt-softmax over L per batch, in place in ws ------------
// Stores sqrt(softmax(norms)) so the GEMM can fold the weight into BOTH
// operands, keeping all f16-split values well inside f16 normal range.
__global__ void softmax_kernel(float* __restrict__ w) {
  __shared__ float red[256];
  const int b = blockIdx.x, t = threadIdx.x;
  float* wb = w + (size_t)b * L_;
  float nv[32];
  float mx = -3.402823466e38f;
  #pragma unroll
  for (int j = 0; j < 32; ++j) {
    nv[j] = wb[t + j * 256];
    mx = fmaxf(mx, nv[j]);
  }
  red[t] = mx; __syncthreads();
  for (int s = 128; s > 0; s >>= 1) {
    if (t < s) red[t] = fmaxf(red[t], red[t + s]);
    __syncthreads();
  }
  mx = red[0]; __syncthreads();
  float sum = 0.f;
  #pragma unroll
  for (int j = 0; j < 32; ++j) { nv[j] = __expf(nv[j] - mx); sum += nv[j]; }
  red[t] = sum; __syncthreads();
  for (int s = 128; s > 0; s >>= 1) {
    if (t < s) red[t] += red[t + s];
    __syncthreads();
  }
  float rinv = __frsqrt_rn(red[0]);               // 1/sqrt(sum)
  #pragma unroll
  for (int j = 0; j < 32; ++j) wb[t + j * 256] = __fsqrt_rn(nv[j]) * rinv;
}

// ------- Kernel 3: out[b] = (SX)^T (SX), S=diag(sqrt(w)); split-f16 WMMA ----
__global__ void __launch_bounds__(256)
gemm_kernel(const float* __restrict__ x, const float* __restrict__ w,
            float* __restrict__ out) {
  // K-contiguous f16 hi/lo slabs; 80B row stride keeps b128 alignment.
  __shared__ __align__(16) ush Ah[128][LDK];
  __shared__ __align__(16) ush Al[128][LDK];
  __shared__ __align__(16) ush Bh[64][LDK];
  __shared__ __align__(16) ush Bl[64][LDK];

  const int b  = blockIdx.z;
  const int D0 = blockIdx.y * 128;
  const int E0 = blockIdx.x * 64;
  const int t  = threadIdx.x;

  const float* xb = x + (size_t)b * L_ * D_;
  const float* wb = w + (size_t)b * L_;

  const int lane  = t & 31;
  const int wid   = t >> 5;
  const int d_off = (wid >> 1) * 32;   // 4 row-blocks of 32
  const int e_off = (wid & 1) * 32;    // 2 col-blocks of 32
  const int hsel  = lane >> 4;         // lane half (ISA fragment layout)
  const int mn    = lane & 15;

  // A(16x32) frag: lane half h, row m: elems 0..7 = K[8h..], 8..15 = K[16+8h..]
  auto ldA = [&](const ush (*arr)[LDK], int d) -> v16h {
    V16 u;
    u.h[0] = *(const v8h*)&arr[d][hsel * 8];
    u.h[1] = *(const v8h*)&arr[d][16 + hsel * 8];
    return u.v;
  };
  // B(32x16) frag: lane half h, col n: elems 0..7 = K[8h..], 8..15 = K[16+8h..]
  auto ldB = [&](const ush (*arr)[LDK], int e) -> v16h {
    V16 u;
    u.h[0] = *(const v8h*)&arr[e][hsel * 8];
    u.h[1] = *(const v8h*)&arr[e][16 + hsel * 8];
    return u.v;
  };

  v8f acc00 = {}, acc01 = {}, acc10 = {}, acc11 = {};
  float fa0[8], fa1[8], fb0[4], fb1[4];  // adjacent-k pairs for pk converts

  // A slab 128d x 32k: 8 pairs/thread; B slab 64e x 32k: 4 pairs/thread.
  // Both sides pre-scaled by sqrt(w)[k].
  auto load_slabs = [&](int k0) {
    #pragma unroll
    for (int i = 0; i < 8; ++i) {
      int pp = i * 256 + t;
      int d  = pp & 127;
      int k  = (pp >> 7) << 1;
      const float* r0 = xb + (size_t)(k0 + k) * D_;
      fa0[i] = r0[D0 + d] * wb[k0 + k];
      fa1[i] = r0[D_ + D0 + d] * wb[k0 + k + 1];
    }
    #pragma unroll
    for (int i = 0; i < 4; ++i) {
      int pp = i * 256 + t;
      int e  = pp & 63;
      int k  = (pp >> 6) << 1;
      const float* r0 = xb + (size_t)(k0 + k) * D_;
      fb0[i] = r0[E0 + e] * wb[k0 + k];
      fb1[i] = r0[D_ + E0 + e] * wb[k0 + k + 1];
    }
  };

  load_slabs(0);

  for (int k0 = 0; k0 < L_; k0 += BK) {
    // Split fp32 -> f16 hi/lo (packed pairs, v_cvt_pk_rtz) and stage to LDS.
    #pragma unroll
    for (int i = 0; i < 8; ++i) {
      int pp = i * 256 + t;
      int d  = pp & 127;
      int k  = (pp >> 7) << 1;
      unsigned h = pk_f16(fa0[i], fa1[i]);
      v2h hv = __builtin_bit_cast(v2h, h);
      *(unsigned*)&Ah[d][k] = h;
      *(unsigned*)&Al[d][k] = pk_f16(fa0[i] - (float)hv.x, fa1[i] - (float)hv.y);
    }
    #pragma unroll
    for (int i = 0; i < 4; ++i) {
      int pp = i * 256 + t;
      int e  = pp & 63;
      int k  = (pp >> 6) << 1;
      unsigned h = pk_f16(fb0[i], fb1[i]);
      v2h hv = __builtin_bit_cast(v2h, h);
      *(unsigned*)&Bh[e][k] = h;
      *(unsigned*)&Bl[e][k] = pk_f16(fb0[i] - (float)hv.x, fb1[i] - (float)hv.y);
    }
    __syncthreads();

    // Pipeline: fetch next slab into registers while WMMAs run on this one.
    if (k0 + BK < L_) {
      load_slabs(k0 + BK);
      int kq = k0 + 2 * BK + (t >> 7);
      if (kq < L_) {
        __builtin_prefetch(xb + (size_t)kq * D_ + D0 + (t & 127), 0, 3);
        __builtin_prefetch(xb + (size_t)kq * D_ + E0 + (t & 63), 0, 3);
      }
    }

    v16h ah0 = ldA(Ah, d_off + mn);
    v16h ah1 = ldA(Ah, d_off + 16 + mn);
    v16h al0 = ldA(Al, d_off + mn);
    v16h al1 = ldA(Al, d_off + 16 + mn);
    v16h bh0 = ldB(Bh, e_off + mn);
    v16h bh1 = ldB(Bh, e_off + 16 + mn);
    v16h bl0 = ldB(Bl, e_off + mn);
    v16h bl1 = ldB(Bl, e_off + 16 + mn);

    // fp32-accurate product: hi*hi + hi*lo + lo*hi (lo*lo ~ 2^-22, dropped)
    acc00 = wmma_f16(ah0, bh0, acc00);
    acc00 = wmma_f16(ah0, bl0, acc00);
    acc00 = wmma_f16(al0, bh0, acc00);
    acc01 = wmma_f16(ah0, bh1, acc01);
    acc01 = wmma_f16(ah0, bl1, acc01);
    acc01 = wmma_f16(al0, bh1, acc01);
    acc10 = wmma_f16(ah1, bh0, acc10);
    acc10 = wmma_f16(ah1, bl0, acc10);
    acc10 = wmma_f16(al1, bh0, acc10);
    acc11 = wmma_f16(ah1, bh1, acc11);
    acc11 = wmma_f16(ah1, bl1, acc11);
    acc11 = wmma_f16(al1, bh1, acc11);

    __syncthreads();
  }

  // C/D layout: VGPR j -> M = 8*hsel + j, N = lane&15.
  float* ob = out + (size_t)b * D_ * D_;
  const int m0 = D0 + d_off + hsel * 8;
  const int n0 = E0 + e_off + mn;
  #pragma unroll
  for (int j = 0; j < 8; ++j) {
    ob[(size_t)(m0 + j)      * D_ + n0]      = acc00[j];
    ob[(size_t)(m0 + j)      * D_ + n0 + 16] = acc01[j];
    ob[(size_t)(m0 + 16 + j) * D_ + n0]      = acc10[j];
    ob[(size_t)(m0 + 16 + j) * D_ + n0 + 16] = acc11[j];
  }
}

extern "C" void kernel_launch(void* const* d_in, const int* in_sizes, int n_in,
                              void* d_out, int out_size, void* d_ws, size_t ws_size,
                              hipStream_t stream) {
  const float* x = (const float*)d_in[0];
  float* out = (float*)d_out;
  float* wts = (float*)d_ws;   // B_*L_ floats (512 KB): norms -> sqrt-softmax

  norms_kernel<<<(B_ * L_) / 8, 256, 0, stream>>>(x, wts);
  softmax_kernel<<<B_, 256, 0, stream>>>(wts);

  dim3 grid(D_ / 64, D_ / 128, B_);       // (4, 2, 16) = 128 workgroups
  gemm_kernel<<<grid, 256, 0, stream>>>(x, wts, out);
}
